// AcousticModel_72000831750704
// MI455X (gfx1250) — matrix operand order repaired and checked by reading, not compiled, not run on hardware
//
#include <hip/hip_runtime.h>

// ---------------------------------------------------------------------------
// CDNA5 (gfx1250) implementation of the Tacotron-style acoustic model.
// Dense math: V_WMMA_F32_16X16X32_F16 (f16 in, f32 accum).
// Tile staging: TENSOR_LOAD_TO_LDS (TDM, D# descriptors, TENSORcnt) for the
// big GEMMs; GLOBAL_LOAD_ASYNC_TO_LDS_B128 (ASYNCcnt) for the batched GEMM.
// Recurrences: input GEMMs hoisted; per-step kernels = h@Wh WMMA + pointwise.
// ---------------------------------------------------------------------------

typedef _Float16 f16_t;
typedef unsigned int u32;
typedef unsigned long long u64;
typedef __attribute__((ext_vector_type(16))) _Float16 v16h;
typedef __attribute__((ext_vector_type(8)))  _Float16 v8h;
typedef __attribute__((ext_vector_type(8)))  float    v8f;
typedef __attribute__((ext_vector_type(4)))  u32      v4u;
typedef __attribute__((ext_vector_type(8)))  int      v8i;
typedef __attribute__((ext_vector_type(4)))  int      v4i;

__device__ inline v8f wmma_f16(v16h a, v16h b, v8f c) {
  return __builtin_amdgcn_wmma_f32_16x16x32_f16(false, a, false, b, (short)0, c,
                                                false, false);
}
// A-fragment: two 8-half chunks at +0 and +16 (K = {hid*8..+7, 16+hid*8..+7})
__device__ inline v16h frag_gap16(const f16_t* p) {
  v16h r;
  *(v8h*)&r = *(const v8h*)p;
  *(((v8h*)&r) + 1) = *(const v8h*)(p + 16);
  return r;
}
// B-fragment: column n, K = {hid*8..+7, 16+hid*8..+7} gathered with row stride ld
__device__ inline v16h frag_cont16(const f16_t* p, int ld) {
  v16h r;
#pragma unroll
  for (int s = 0; s < 8; ++s) {
    r[s] = p[(size_t)s * ld];
    r[s + 8] = p[(size_t)(16 + s) * ld];
  }
  return r;
}
__device__ inline float sigm(float x) { return 1.0f / (1.0f + __expf(-x)); }

// gfx1250 async global->LDS copy (16B per lane), tracked by ASYNCcnt.
__device__ inline void async_g2l_b128(unsigned lds_off, const f16_t* g) {
  asm volatile("global_load_async_to_lds_b128 %0, %1, off"
               :: "v"(lds_off), "v"(g) : "memory");
}
#define WAIT_ASYNC_LE(imm) asm volatile("s_wait_asynccnt " #imm ::: "memory")

// ---- Tensor Data Mover: one 2D f16 tile global->LDS (D# per ISA 8.3/8.4).
// elems_row  = valid elements per row from tile start (OOB cols read as 0)
// rows       = valid rows from tile start
// tile_w/h   = tile dims (elements / rows)
// stride     = row stride in elements
// pad codes  : interval 0=2,1=4,2=8,3=16,4=32,5=64 dwords; amount n -> n+1 dw
// NOTE: this toolchain's builtin is the 6-arg form
// (uint32x4 g0, int32x8 g1, int32x4 g2, int32x4 g3, int32x8 g4, i32 cpol).
__device__ inline void tdm_load_2d(u32 lds_addr, const void* gaddr,
                                   u32 elems_row, u32 rows, u32 tile_w,
                                   u32 tile_h, u64 stride, u32 pad_int,
                                   u32 pad_amt) {
  u64 ga = (u64)(size_t)gaddr;
  v4u g0;
  g0[0] = 1u;                                            // count=1, user mode
  g0[1] = lds_addr;                                      // lds_addr [63:32]
  g0[2] = (u32)ga;                                       // global_addr low
  g0[3] = (u32)((ga >> 32) & 0x1FFFFFFu) | (2u << 30);   // addr[56:32]|type=2
  v8i g1;
  g1[0] = (int)((1u << 16) |                 // data_size = 2 bytes
                (1u << 20) |                 // pad_enable
                (pad_int << 22) | (pad_amt << 25));
  g1[1] = (int)((elems_row & 0xFFFFu) << 16);            // tensor_dim0 lo16
  g1[2] = (int)(((elems_row >> 16) & 0xFFFFu) |          // tensor_dim0 hi16
                ((rows & 0xFFFFu) << 16));               // tensor_dim1 lo16
  g1[3] = (int)(((rows >> 16) & 0xFFFFu) |               // tensor_dim1 hi16
                ((tile_w & 0xFFFFu) << 16));             // tile_dim0
  g1[4] = (int)(tile_h & 0xFFFFu);                       // tile_dim1
  g1[5] = (int)(u32)(stride & 0xFFFFFFFFu);              // dim0_stride lo32
  g1[6] = (int)(u32)((stride >> 32) & 0xFFFFu);          // dim0_stride hi16
  g1[7] = 0;
  v4i gz = {0, 0, 0, 0};
  v8i gz8 = {0, 0, 0, 0, 0, 0, 0, 0};
  __builtin_amdgcn_tensor_load_to_lds(g0, g1, gz, gz, gz8, 0);
}

// ============================ GEMM tiling ==================================
// C[M,N] = A[M,K] * B[K,N] (+bias). A f16 lda=KP (K mult of 32, zero padded),
// B f16 packed [KP,N]. Block tile 128x128, 8 waves (4x2 wavetile 32x64).
// Contract: M is a multiple of 128 at every call site (true for this model).
#define GBM 128
#define GBN 128

// ---- variant 1: TDM-staged (wave 0 issues 2 tensor loads/tile, TENSORcnt),
// double-buffered LDS; other waves synchronize at the workgroup barrier.
__global__ __launch_bounds__(256) void gemm_f16_tdm_k(
    const f16_t* __restrict__ A, int lda,
    const f16_t* __restrict__ Bm, int ldb,
    void* __restrict__ Cv, int ldc,
    int M, int N, int K, const float* __restrict__ bias, int out16) {
  __shared__ f16_t As[2][GBM][40];      // 32 K + 8 pad halfs (TDM pad 16dw+4dw)
  __shared__ f16_t Bs[2][32][GBN + 8];  // 128 N + 8 pad halfs (TDM pad 64dw+4dw)
  const int tid = threadIdx.x;
  const int lane = tid & 31, wid = tid >> 5;
  const int wm = wid >> 1, wn = wid & 1;
  const int lr = lane & 15, lh = lane >> 4;
  const int m0 = blockIdx.y * GBM, n0 = blockIdx.x * GBN;
  const u32 bn = (u32)((N - n0 < GBN) ? (N - n0) : GBN);  // valid B cols

  const v8f vz = {0.f, 0.f, 0.f, 0.f, 0.f, 0.f, 0.f, 0.f};
  v8f acc[2][4];
  for (int i = 0; i < 2; ++i)
    for (int j = 0; j < 4; ++j) acc[i][j] = vz;

  const int nkb = K >> 5;
  if (wid == 0) {
    tdm_load_2d((u32)(size_t)&As[0][0][0], A + (size_t)m0 * lda,
                32, 128, 32, 128, (u64)lda, 3, 3);
    tdm_load_2d((u32)(size_t)&Bs[0][0][0], Bm + n0,
                bn, 32, GBN, 32, (u64)ldb, 5, 3);
  }
  for (int kb = 0; kb < nkb; ++kb) {
    const int cur = kb & 1;
    if (wid == 0) {
      if (kb + 1 < nkb) {
        tdm_load_2d((u32)(size_t)&As[cur ^ 1][0][0],
                    A + (size_t)m0 * lda + (kb + 1) * 32,
                    32, 128, 32, 128, (u64)lda, 3, 3);
        tdm_load_2d((u32)(size_t)&Bs[cur ^ 1][0][0],
                    Bm + (size_t)((kb + 1) * 32) * ldb + n0,
                    bn, 32, GBN, 32, (u64)ldb, 5, 3);
        __builtin_amdgcn_s_wait_tensorcnt((short)2);  // tile kb done (in order)
      } else {
        __builtin_amdgcn_s_wait_tensorcnt((short)0);
      }
    }
    __syncthreads();
    v16h afr[2], bfr[4];
#pragma unroll
    for (int mi = 0; mi < 2; ++mi)
      afr[mi] = frag_gap16(&As[cur][wm * 32 + mi * 16 + lr][lh * 8]);
#pragma unroll
    for (int ni = 0; ni < 4; ++ni)
      bfr[ni] = frag_cont16(&Bs[cur][lh * 8][wn * 64 + ni * 16 + lr], GBN + 8);
#pragma unroll
    for (int mi = 0; mi < 2; ++mi)
#pragma unroll
      for (int ni = 0; ni < 4; ++ni)
        acc[mi][ni] = wmma_f16(afr[mi], bfr[ni], acc[mi][ni]);
    __syncthreads();   // consumers done before this buffer is overwritten
  }
  for (int mi = 0; mi < 2; ++mi)
    for (int ni = 0; ni < 4; ++ni)
      for (int i = 0; i < 8; ++i) {
        int row = m0 + wm * 32 + mi * 16 + i + 8 * lh;
        int col = n0 + wn * 64 + ni * 16 + lr;
        if (row < M && col < N) {
          float v = acc[mi][ni][i];
          if (bias) v += bias[col];
          size_t off = (size_t)row * ldc + col;
          if (out16) ((f16_t*)Cv)[off] = (f16_t)v;
          else       ((float*)Cv)[off] = v;
        }
      }
}

// ---- variant 2: per-lane async-to-LDS staging (ASYNCcnt), batched (z dim).
__global__ __launch_bounds__(256) void gemm_f16_k(
    const f16_t* __restrict__ A, int lda, long long sA,
    const f16_t* __restrict__ Bm, int ldb, long long sB,
    void* __restrict__ Cv, int ldc, long long sC,
    int M, int N, int K, const float* __restrict__ bias, int out16) {
  __shared__ f16_t As[2][GBM][40];
  __shared__ f16_t Bs[2][32][GBN + 8];
  const int tid = threadIdx.x;
  const int lane = tid & 31, wid = tid >> 5;
  const int wm = wid >> 1, wn = wid & 1;
  const int lr = lane & 15, lh = lane >> 4;
  const int m0 = blockIdx.y * GBM, n0 = blockIdx.x * GBN;
  const f16_t* Ab = A + (size_t)blockIdx.z * sA;
  const f16_t* Bb = Bm + (size_t)blockIdx.z * sB;

  const v8f vz = {0.f, 0.f, 0.f, 0.f, 0.f, 0.f, 0.f, 0.f};
  const v8h hz = {0, 0, 0, 0, 0, 0, 0, 0};
  v8f acc[2][4];
  for (int i = 0; i < 2; ++i)
    for (int j = 0; j < 4; ++j) acc[i][j] = vz;

  auto issue = [&](int kb, int buf) {
#pragma unroll
    for (int p = 0; p < 2; ++p) {
      int c = tid + p * 256;
      int row = c >> 2, seg = c & 3;
      async_g2l_b128((unsigned)(size_t)&As[buf][row][seg * 8],
                     Ab + (size_t)(m0 + row) * lda + kb * 32 + seg * 8);
      int br = c >> 4, bseg = c & 15;
      int n = n0 + bseg * 8;
      if (n < N) {
        async_g2l_b128((unsigned)(size_t)&Bs[buf][br][bseg * 8],
                       Bb + (size_t)(kb * 32 + br) * ldb + n);
      } else {
        *(v8h*)&Bs[buf][br][bseg * 8] = hz;
      }
    }
  };

  const int nkb = K >> 5;
  issue(0, 0);
  for (int kb = 0; kb < nkb; ++kb) {
    const int cur = kb & 1;
    if (kb + 1 < nkb) {
      issue(kb + 1, cur ^ 1);
      WAIT_ASYNC_LE(0x4);
    } else {
      WAIT_ASYNC_LE(0x0);
    }
    __syncthreads();
    v16h afr[2], bfr[4];
#pragma unroll
    for (int mi = 0; mi < 2; ++mi)
      afr[mi] = frag_gap16(&As[cur][wm * 32 + mi * 16 + lr][lh * 8]);
#pragma unroll
    for (int ni = 0; ni < 4; ++ni)
      bfr[ni] = frag_cont16(&Bs[cur][lh * 8][wn * 64 + ni * 16 + lr], GBN + 8);
#pragma unroll
    for (int mi = 0; mi < 2; ++mi)
#pragma unroll
      for (int ni = 0; ni < 4; ++ni)
        acc[mi][ni] = wmma_f16(afr[mi], bfr[ni], acc[mi][ni]);
    __syncthreads();
  }
  for (int mi = 0; mi < 2; ++mi)
    for (int ni = 0; ni < 4; ++ni)
      for (int i = 0; i < 8; ++i) {
        int row = m0 + wm * 32 + mi * 16 + i + 8 * lh;
        int col = n0 + wn * 64 + ni * 16 + lr;
        if (row < M && col < N) {
          float v = acc[mi][ni][i];
          if (bias) v += bias[col];
          size_t off = (size_t)row * ldc + col + (size_t)blockIdx.z * sC;
          if (out16) ((f16_t*)Cv)[off] = (f16_t)v;
          else       ((float*)Cv)[off] = v;
        }
      }
}

// ===================== encoder bi-LSTM per-step kernel =====================
__global__ __launch_bounds__(256) void lstm_step_enc_k(
    const f16_t* __restrict__ Wh, const f16_t* __restrict__ gx,
    const float* __restrict__ bias, f16_t* __restrict__ h16,
    float* __restrict__ c_st, f16_t* __restrict__ enc16,
    const int* __restrict__ lengths, int t, int L, int dir) {
  extern __shared__ float gsm[];
  const int tid = threadIdx.x, lane = tid & 31, wid = tid >> 5;
  const int lr = lane & 15, lh = lane >> 4;
  const v8f vz = {0.f, 0.f, 0.f, 0.f, 0.f, 0.f, 0.f, 0.f};
  v8f acc[8];
  for (int i = 0; i < 8; ++i) acc[i] = vz;
  for (int kk = 0; kk < 8; ++kk) {
    if (kk + 1 < 8)
      __builtin_prefetch(Wh + (size_t)((kk + 1) * 32 + lane) * 1024 + wid * 128, 0, 3);
    v16h a = frag_gap16(h16 + lr * 256 + kk * 32 + lh * 8);
#pragma unroll
    for (int ni = 0; ni < 8; ++ni) {
      v16h b = frag_cont16(Wh + (size_t)(kk * 32 + lh * 8) * 1024 + wid * 128 + ni * 16 + lr, 1024);
      acc[ni] = wmma_f16(a, b, acc[ni]);
    }
  }
  const int gl = dir ? (L - 1 - t) : t;
  for (int ni = 0; ni < 8; ++ni)
    for (int i = 0; i < 8; ++i) {
      int b = i + 8 * lh;
      int col = wid * 128 + ni * 16 + lr;
      gsm[b * 1024 + col] =
          acc[ni][i] + (float)gx[((size_t)b * L + gl) * 1024 + col] + bias[col];
    }
  __syncthreads();
  const int b = tid >> 4, e0 = (tid & 15) * 16;
  for (int q = 0; q < 16; ++q) {
    int e = e0 + q;
    float ig = gsm[b * 1024 + e];
    float gg = gsm[b * 1024 + 256 + e];
    float ff = gsm[b * 1024 + 512 + e];
    float oo = gsm[b * 1024 + 768 + e];
    float co = c_st[b * 256 + e];
    float cn = sigm(ff + 1.f) * co + sigm(ig) * tanhf(gg);
    float hn = sigm(oo) * tanhf(cn);
    enc16[((size_t)b * L + gl) * 512 + dir * 256 + e] = (f16_t)hn;
    if (dir && t < L - 1 && (L - 2 - t) >= (lengths[b] - 1)) { cn = 0.f; hn = 0.f; }
    c_st[b * 256 + e] = cn;
    h16[b * 256 + e] = (f16_t)hn;
  }
}

// ===================== decoder 2-layer LSTM per-step =======================
__global__ __launch_bounds__(512) void lstm_step_dec_k(
    const f16_t* __restrict__ Wh1, const f16_t* __restrict__ Wh2,
    const f16_t* __restrict__ gx1, const f16_t* __restrict__ gx2,
    const float* __restrict__ b1, const float* __restrict__ b2,
    f16_t* __restrict__ h1_16, float* __restrict__ h1f, float* __restrict__ c1,
    float* __restrict__ h2f, float* __restrict__ c2,
    f16_t* __restrict__ proj_in,
    const unsigned char* __restrict__ mh1, const unsigned char* __restrict__ mc1,
    const unsigned char* __restrict__ mh2, const unsigned char* __restrict__ mc2,
    int t, int T) {
  extern __shared__ char smem[];
  float* gsm = (float*)smem;                      // 16*2048 f32
  f16_t* hcat = (f16_t*)(smem + 16 * 2048 * 4);   // 16*1024 f16
  const int tid = threadIdx.x, lane = tid & 31, wid = tid >> 5;
  const int lr = lane & 15, lh = lane >> 4;
  const v8f vz = {0.f, 0.f, 0.f, 0.f, 0.f, 0.f, 0.f, 0.f};
  v8f acc[8];
  for (int i = 0; i < 8; ++i) acc[i] = vz;
  for (int kk = 0; kk < 16; ++kk) {
    if (kk + 1 < 16)
      __builtin_prefetch(Wh1 + (size_t)((kk + 1) * 32 + lane) * 2048 + wid * 128, 0, 3);
    v16h a = frag_gap16(h1_16 + lr * 512 + kk * 32 + lh * 8);
#pragma unroll
    for (int ni = 0; ni < 8; ++ni) {
      v16h b = frag_cont16(Wh1 + (size_t)(kk * 32 + lh * 8) * 2048 + wid * 128 + ni * 16 + lr, 2048);
      acc[ni] = wmma_f16(a, b, acc[ni]);
    }
  }
  for (int ni = 0; ni < 8; ++ni)
    for (int i = 0; i < 8; ++i) {
      int b = i + 8 * lh;
      int col = wid * 128 + ni * 16 + lr;
      gsm[b * 2048 + col] =
          acc[ni][i] + (float)gx1[((size_t)b * T + t) * 2048 + col] + b1[col];
    }
  __syncthreads();
  {
    const int b = tid >> 5, e0 = (tid & 31) * 16;
    for (int q = 0; q < 16; ++q) {
      int e = e0 + q;
      float ig = gsm[b * 2048 + e];
      float gg = gsm[b * 2048 + 512 + e];
      float ff = gsm[b * 2048 + 1024 + e];
      float oo = gsm[b * 2048 + 1536 + e];
      float co = c1[b * 512 + e];
      float cn = sigm(ff + 1.f) * co + sigm(ig) * tanhf(gg);
      float nh = sigm(oo) * tanhf(cn);
      size_t bt = (size_t)b * T + t;
      proj_in[bt * 1536 + 512 + e] = (f16_t)nh;
      hcat[b * 1024 + e] = (f16_t)nh;
      hcat[b * 1024 + 512 + e] = (f16_t)h2f[b * 512 + e];
      size_t mo = bt * 512 + e;
      float ho = h1f[b * 512 + e];
      float hs = mh1[mo] ? ho : nh;
      float cs = mc1[mo] ? co : cn;
      h1f[b * 512 + e] = hs;
      c1[b * 512 + e] = cs;
      h1_16[b * 512 + e] = (f16_t)hs;
    }
  }
  __syncthreads();
  for (int i = 0; i < 8; ++i) acc[i] = vz;
  for (int kk = 0; kk < 32; ++kk) {
    if (kk + 1 < 32)
      __builtin_prefetch(Wh2 + (size_t)((kk + 1) * 32 + lane) * 2048 + wid * 128, 0, 3);
    v16h a = frag_gap16(hcat + lr * 1024 + kk * 32 + lh * 8);
#pragma unroll
    for (int ni = 0; ni < 8; ++ni) {
      v16h b = frag_cont16(Wh2 + (size_t)(kk * 32 + lh * 8) * 2048 + wid * 128 + ni * 16 + lr, 2048);
      acc[ni] = wmma_f16(a, b, acc[ni]);
    }
  }
  for (int ni = 0; ni < 8; ++ni)
    for (int i = 0; i < 8; ++i) {
      int b = i + 8 * lh;
      int col = wid * 128 + ni * 16 + lr;
      gsm[b * 2048 + col] =
          acc[ni][i] + (float)gx2[((size_t)b * T + t) * 2048 + col] + b2[col];
    }
  __syncthreads();
  {
    const int b = tid >> 5, e0 = (tid & 31) * 16;
    for (int q = 0; q < 16; ++q) {
      int e = e0 + q;
      float ig = gsm[b * 2048 + e];
      float gg = gsm[b * 2048 + 512 + e];
      float ff = gsm[b * 2048 + 1024 + e];
      float oo = gsm[b * 2048 + 1536 + e];
      float co = c2[b * 512 + e];
      float cn = sigm(ff + 1.f) * co + sigm(ig) * tanhf(gg);
      float nh = sigm(oo) * tanhf(cn);
      size_t bt = (size_t)b * T + t;
      proj_in[bt * 1536 + 1024 + e] = (f16_t)nh;
      size_t mo = bt * 512 + e;
      float ho = h2f[b * 512 + e];
      h2f[b * 512 + e] = mh2[mo] ? ho : nh;
      c2[b * 512 + e] = mc2[mo] ? co : cn;
    }
  }
}

// ============================== glue kernels ===============================
__global__ void pack_b_k(const float* __restrict__ s, f16_t* __restrict__ d,
                         int K, int N, int KP) {
  int i = blockIdx.x * 256 + threadIdx.x;
  int tot = KP * N;
  if (i >= tot) return;
  int k = i / N, n = i - k * N;
  d[i] = (k < K) ? (f16_t)s[(size_t)k * N + n] : (f16_t)0.f;
}
__global__ void pack_a_k(const float* __restrict__ s, f16_t* __restrict__ d,
                         int R, int C, int KP) {
  long long i = (long long)blockIdx.x * 256 + threadIdx.x;
  long long tot = (long long)R * KP;
  if (i >= tot) return;
  int k = (int)(i % KP);
  long long r = i / KP;
  d[i] = (k < C) ? (f16_t)s[r * C + k] : (f16_t)0.f;
}
__global__ void copy_h_k(const float* __restrict__ s, f16_t* __restrict__ d,
                         int R, int C, int ldd, int coff) {
  long long i = (long long)blockIdx.x * 256 + threadIdx.x;
  long long tot = (long long)R * C;
  if (i >= tot) return;
  long long r = i / C;
  int c = (int)(i % C);
  d[r * ldd + coff + c] = (f16_t)s[i];
}
__global__ void embed_k(const int* __restrict__ ph, const float* __restrict__ em,
                        float* __restrict__ x, int n, int E) {
  int i = blockIdx.x * 256 + threadIdx.x;
  if (i >= n) return;
  int row = i / E, e = i - row * E;
  x[i] = em[(size_t)ph[row] * E + e];
}
__global__ void im2col_k(const float* __restrict__ s, f16_t* __restrict__ d,
                         int Bn, int Ln, int C, int ks, int KP) {
  long long i = (long long)blockIdx.x * 256 + threadIdx.x;
  long long tot = (long long)Bn * Ln * KP;
  if (i >= tot) return;
  int kcol = (int)(i % KP);
  long long row = i / KP;
  int l = (int)(row % Ln);
  int b = (int)(row / Ln);
  f16_t v = (f16_t)0.f;
  if (kcol < ks * C) {
    int kk = kcol / C, ci = kcol - kk * C;
    int ls = l + kk - ks / 2;
    if (ls >= 0 && ls < Ln) v = (f16_t)s[((size_t)b * Ln + ls) * C + ci];
  }
  d[i] = v;
}
__global__ void bn_stats_k(const float* __restrict__ x, float* __restrict__ mean,
                           float* __restrict__ var, int R, int C) {
  __shared__ float rs[256], rs2[256];
  int c = blockIdx.x, tid = threadIdx.x;
  float s = 0.f, s2 = 0.f;
  for (int r = tid; r < R; r += 256) {
    float v = x[(size_t)r * C + c];
    s += v; s2 += v * v;
  }
  rs[tid] = s; rs2[tid] = s2;
  __syncthreads();
  for (int o = 128; o > 0; o >>= 1) {
    if (tid < o) { rs[tid] += rs[tid + o]; rs2[tid] += rs2[tid + o]; }
    __syncthreads();
  }
  if (tid == 0) {
    float m = rs[0] / R;
    mean[c] = m;
    var[c] = rs2[0] / R - m * m;
  }
}
__global__ void bn_act_k(const float* __restrict__ x, float* __restrict__ y,
                         const float* __restrict__ mean, const float* __restrict__ var,
                         const float* __restrict__ sc, const float* __restrict__ of,
                         int R, int C, int act) {
  long long i = (long long)blockIdx.x * 256 + threadIdx.x;
  long long tot = (long long)R * C;
  if (i >= tot) return;
  int c = (int)(i % C);
  float v = (x[i] - mean[c]) * rsqrtf(var[c] + 1e-5f) * sc[c] + of[c];
  if (act == 1) v = fmaxf(v, 0.f);
  else if (act == 2) v = tanhf(v);
  y[i] = v;
}
__global__ void cumsum_k(const float* __restrict__ dur, float* __restrict__ midp,
                         int Bn, int L) {
  int b = blockIdx.x * blockDim.x + threadIdx.x;
  if (b >= Bn) return;
  float run = 0.f;
  for (int l = 0; l < L; ++l) {
    float d = dur[b * L + l];
    run += d;
    midp[b * L + l] = run - 0.5f * d;
  }
}
__global__ void ups_softmax_k(const float* __restrict__ midp,
                              f16_t* __restrict__ w16, int T, int L) {
  __shared__ float red[256];
  int bt = blockIdx.x;
  int b = bt / T, t = bt - b * T;
  int l = threadIdx.x;
  float d = midp[b * L + l] - (float)t;
  float s = -d * d * 0.1f;
  red[l] = s;
  __syncthreads();
  for (int o = 128; o > 0; o >>= 1) {
    if (l < o) red[l] = fmaxf(red[l], red[l + o]);
    __syncthreads();
  }
  float mx = red[0];
  __syncthreads();
  float e = __expf(s - mx);
  red[l] = e;
  __syncthreads();
  for (int o = 128; o > 0; o >>= 1) {
    if (l < o) red[l] += red[l + o];
    __syncthreads();
  }
  w16[((size_t)b * T + t) * L + l] = (f16_t)(e / red[0]);
}
__global__ void add_out_k(const float* __restrict__ mel, const float* __restrict__ res,
                          const float* __restrict__ b4, float* __restrict__ out,
                          int n, int C) {
  int i = blockIdx.x * 256 + threadIdx.x;
  if (i >= n) return;
  out[i] = mel[i] + res[i] + b4[i % C];
}

// ============================== host driver ================================
extern "C" void kernel_launch(void* const* d_in, const int* in_sizes, int n_in,
                              void* d_out, int out_size, void* d_ws, size_t ws_size,
                              hipStream_t stream) {
  (void)in_sizes; (void)n_in; (void)out_size; (void)ws_size;
  const int B = 16, L = 256, T = 512, E = 256, MEL = 80;
  const int BL = B * L, BT = B * T;

  const int*   phon = (const int*)d_in[0];
  const int*   lens = (const int*)d_in[1];
  const float* mels = (const float*)d_in[2];
  const float* durs = (const float*)d_in[3];
  const unsigned char* zh1 = (const unsigned char*)d_in[4];
  const unsigned char* zc1 = (const unsigned char*)d_in[5];
  const unsigned char* zh2 = (const unsigned char*)d_in[6];
  const unsigned char* zc2 = (const unsigned char*)d_in[7];
  const float* embed = (const float*)d_in[8];
  const float* ecw = (const float*)d_in[9];
  const float* ebs = (const float*)d_in[10];
  const float* ebo = (const float*)d_in[11];
  const float* lfw = (const float*)d_in[12];
  const float* lfb = (const float*)d_in[13];
  const float* lbw = (const float*)d_in[14];
  const float* lbb = (const float*)d_in[15];
  const float* pw1 = (const float*)d_in[16];
  const float* p1s = (const float*)d_in[17];
  const float* p1o = (const float*)d_in[18];
  const float* pw2 = (const float*)d_in[19];
  const float* p2s = (const float*)d_in[20];
  const float* p2o = (const float*)d_in[21];
  const float* dw1 = (const float*)d_in[22];
  const float* db1 = (const float*)d_in[23];
  const float* dw2 = (const float*)d_in[24];
  const float* db2 = (const float*)d_in[25];
  const float* prw = (const float*)d_in[26];
  const float* prb = (const float*)d_in[27];
  const float* pow0 = (const float*)d_in[28];
  const float* powm = (const float*)d_in[29];
  const float* pow4 = (const float*)d_in[30];
  const float* pob4 = (const float*)d_in[31];
  const float* pobs = (const float*)d_in[32];
  const float* pobo = (const float*)d_in[33];

  char* base = (char*)d_ws;
  size_t pos = 0;
  auto alloc = [&](size_t bytes) -> void* {
    pos = (pos + 255) & ~(size_t)255;
    void* p = base + pos;
    pos += bytes;
    return p;
  };

  f16_t* encW[3];
  for (int i = 0; i < 3; ++i) encW[i] = (f16_t*)alloc((size_t)768 * 256 * 2);
  f16_t* Wxf = (f16_t*)alloc((size_t)256 * 1024 * 2);
  f16_t* Whf = (f16_t*)alloc((size_t)256 * 1024 * 2);
  f16_t* Wxb = (f16_t*)alloc((size_t)256 * 1024 * 2);
  f16_t* Whb = (f16_t*)alloc((size_t)256 * 1024 * 2);
  f16_t* Pre1 = (f16_t*)alloc((size_t)96 * 256 * 2);
  f16_t* Pre2 = (f16_t*)alloc((size_t)256 * 256 * 2);
  f16_t* W1x = (f16_t*)alloc((size_t)768 * 2048 * 2);
  f16_t* W1h = (f16_t*)alloc((size_t)512 * 2048 * 2);
  f16_t* W2x = (f16_t*)alloc((size_t)768 * 2048 * 2);
  f16_t* W2h = (f16_t*)alloc((size_t)1024 * 2048 * 2);
  f16_t* Projw = (f16_t*)alloc((size_t)1536 * 80 * 2);
  f16_t* Post0 = (f16_t*)alloc((size_t)416 * 512 * 2);
  f16_t* Postm[3];
  for (int i = 0; i < 3; ++i) Postm[i] = (f16_t*)alloc((size_t)2560 * 512 * 2);
  f16_t* Post4 = (f16_t*)alloc((size_t)2560 * 80 * 2);
  float* encA = (float*)alloc((size_t)BL * 256 * 4);
  float* encB = (float*)alloc((size_t)BL * 256 * 4);
  f16_t* A16 = (f16_t*)alloc((size_t)BT * 2560 * 2);
  f16_t* gxf = (f16_t*)alloc((size_t)BL * 1024 * 2);
  f16_t* gxb = (f16_t*)alloc((size_t)BL * 1024 * 2);
  f16_t* enc16 = (f16_t*)alloc((size_t)BL * 512 * 2);
  float* midp = (float*)alloc((size_t)B * L * 4);
  f16_t* w16 = (f16_t*)alloc((size_t)B * T * L * 2);
  float* cond = (float*)alloc((size_t)BT * 512 * 4);
  float* ppA = (float*)alloc((size_t)BT * 512 * 4);
  float* ppB = (float*)alloc((size_t)BT * 512 * 4);
  f16_t* gx1 = (f16_t*)alloc((size_t)BT * 2048 * 2);
  f16_t* gx2 = (f16_t*)alloc((size_t)BT * 2048 * 2);
  f16_t* din16 = (f16_t*)alloc((size_t)BT * 768 * 2);
  f16_t* pin16 = (f16_t*)alloc((size_t)BT * 1536 * 2);
  float* resb = (float*)alloc((size_t)BT * 80 * 4);
  float* mean = (float*)alloc(512 * 4);
  float* var = (float*)alloc(512 * 4);
  f16_t* hf16 = (f16_t*)alloc(16 * 256 * 2);
  float* cf = (float*)alloc(16 * 256 * 4);
  f16_t* hb16 = (f16_t*)alloc(16 * 256 * 2);
  float* cb = (float*)alloc(16 * 256 * 4);
  f16_t* h1_16 = (f16_t*)alloc(16 * 512 * 2);
  float* h1f = (float*)alloc(16 * 512 * 4);
  float* c1 = (float*)alloc(16 * 512 * 4);
  float* h2f = (float*)alloc(16 * 512 * 4);
  float* c2 = (float*)alloc(16 * 512 * 4);

  auto blks = [](long long n) { return dim3((unsigned)((n + 255) / 256)); };
  auto packB = [&](const float* s, f16_t* d, int K, int N, int KP) {
    pack_b_k<<<blks((long long)KP * N), 256, 0, stream>>>(s, d, K, N, KP);
  };
  // TDM-staged GEMM for all non-batched matmuls
  auto gemmT = [&](const f16_t* A, int lda, const f16_t* Bm, int ldb, void* C,
                   int ldc, int M, int N, int K, const float* bias, int out16) {
    dim3 grid((N + GBN - 1) / GBN, (M + GBM - 1) / GBM, 1);
    gemm_f16_tdm_k<<<grid, 256, 0, stream>>>(A, lda, Bm, ldb, C, ldc,
                                             M, N, K, bias, out16);
  };
  // async-staged batched GEMM (upsampling einsum)
  auto gemmA = [&](const f16_t* A, int lda, long long sA, const f16_t* Bm,
                   int ldb, long long sB, void* C, int ldc, long long sC,
                   int M, int N, int K, const float* bias, int out16, int batch) {
    dim3 grid((N + GBN - 1) / GBN, (M + GBM - 1) / GBM, batch);
    gemm_f16_k<<<grid, 256, 0, stream>>>(A, lda, sA, Bm, ldb, sB, C, ldc, sC,
                                         M, N, K, bias, out16);
  };

  // ---- pack weights (f16, K padded to mult of 32)
  for (int i = 0; i < 3; ++i) packB(ecw + (size_t)i * 768 * 256, encW[i], 768, 256, 768);
  packB(lfw, Wxf, 256, 1024, 256);
  packB(lfw + 256 * 1024, Whf, 256, 1024, 256);
  packB(lbw, Wxb, 256, 1024, 256);
  packB(lbw + 256 * 1024, Whb, 256, 1024, 256);
  packB(pw1, Pre1, 80, 256, 96);
  packB(pw2, Pre2, 256, 256, 256);
  packB(dw1, W1x, 768, 2048, 768);
  packB(dw1 + (size_t)768 * 2048, W1h, 512, 2048, 512);
  packB(dw2, W2x, 768, 2048, 768);
  packB(dw2 + (size_t)768 * 2048, W2h, 1024, 2048, 1024);
  packB(prw, Projw, 1536, 80, 1536);
  packB(pow0, Post0, 400, 512, 416);
  for (int i = 0; i < 3; ++i) packB(powm + (size_t)i * 2560 * 512, Postm[i], 2560, 512, 2560);
  packB(pow4, Post4, 2560, 80, 2560);

  // ---- token encoder: embedding + 3x(conv1d k3 + BN + ReLU)
  embed_k<<<blks((long long)BL * E), 256, 0, stream>>>(phon, embed, encA, BL * E, E);
  for (int i = 0; i < 3; ++i) {
    im2col_k<<<blks((long long)BL * 768), 256, 0, stream>>>(encA, A16, B, L, 256, 3, 768);
    gemmT(A16, 768, encW[i], 256, encB, 256, BL, 256, 768, nullptr, 0);
    bn_stats_k<<<256, 256, 0, stream>>>(encB, mean, var, BL, 256);
    bn_act_k<<<blks((long long)BL * 256), 256, 0, stream>>>(encB, encA, mean, var,
        ebs + i * 256, ebo + i * 256, BL, 256, 1);
  }
  // ---- bi-LSTM: hoisted x@Wx for all steps, then per-step h@Wh kernels
  pack_a_k<<<blks((long long)BL * 256), 256, 0, stream>>>(encA, A16, BL, 256, 256);
  gemmT(A16, 256, Wxf, 1024, gxf, 1024, BL, 1024, 256, nullptr, 1);
  gemmT(A16, 256, Wxb, 1024, gxb, 1024, BL, 1024, 256, nullptr, 1);
  (void)hipMemsetAsync(hf16, 0, 16 * 256 * 2, stream);
  (void)hipMemsetAsync(cf, 0, 16 * 256 * 4, stream);
  (void)hipMemsetAsync(hb16, 0, 16 * 256 * 2, stream);
  (void)hipMemsetAsync(cb, 0, 16 * 256 * 4, stream);
  for (int t = 0; t < L; ++t)
    lstm_step_enc_k<<<1, 256, 16 * 1024 * 4, stream>>>(Whf, gxf, lfb, hf16, cf,
                                                       enc16, lens, t, L, 0);
  for (int t = 0; t < L; ++t)
    lstm_step_enc_k<<<1, 256, 16 * 1024 * 4, stream>>>(Whb, gxb, lbb, hb16, cb,
                                                       enc16, lens, t, L, 1);

  // ---- duration-based Gaussian-softmax upsampling: w @ enc (batched GEMM)
  cumsum_k<<<1, 32, 0, stream>>>(durs, midp, B, L);
  ups_softmax_k<<<BT, 256, 0, stream>>>(midp, w16, T, L);
  gemmA(w16, 256, (long long)T * L, enc16, 512, (long long)L * 512,
        cond, 512, (long long)T * 512, T, 512, 256, nullptr, 0, B);

  // ---- prenet
  pack_a_k<<<blks((long long)BT * 96), 256, 0, stream>>>(mels, A16, BT, 80, 96);
  gemmT(A16, 96, Pre1, 256, ppA, 256, BT, 256, 96, nullptr, 0);
  bn_stats_k<<<256, 256, 0, stream>>>(ppA, mean, var, BT, 256);
  bn_act_k<<<blks((long long)BT * 256), 256, 0, stream>>>(ppA, ppB, mean, var, p1s, p1o, BT, 256, 1);
  pack_a_k<<<blks((long long)BT * 256), 256, 0, stream>>>(ppB, A16, BT, 256, 256);
  gemmT(A16, 256, Pre2, 256, ppA, 256, BT, 256, 256, nullptr, 0);
  bn_stats_k<<<256, 256, 0, stream>>>(ppA, mean, var, BT, 256);
  bn_act_k<<<blks((long long)BT * 256), 256, 0, stream>>>(ppA, ppB, mean, var, p2s, p2o, BT, 256, 1);

  // ---- assemble decoder input / proj input (cond ‖ prenet)
  copy_h_k<<<blks((long long)BT * 512), 256, 0, stream>>>(cond, din16, BT, 512, 768, 0);
  copy_h_k<<<blks((long long)BT * 256), 256, 0, stream>>>(ppB, din16, BT, 256, 768, 512);
  copy_h_k<<<blks((long long)BT * 512), 256, 0, stream>>>(cond, pin16, BT, 512, 1536, 0);

  // ---- decoder: hoisted x@Wx GEMMs, then 512 fused 2-layer step kernels
  gemmT(din16, 768, W1x, 2048, gx1, 2048, BT, 2048, 768, nullptr, 1);
  gemmT(din16, 768, W2x, 2048, gx2, 2048, BT, 2048, 768, nullptr, 1);
  (void)hipMemsetAsync(h1_16, 0, 16 * 512 * 2, stream);
  (void)hipMemsetAsync(h1f, 0, 16 * 512 * 4, stream);
  (void)hipMemsetAsync(c1, 0, 16 * 512 * 4, stream);
  (void)hipMemsetAsync(h2f, 0, 16 * 512 * 4, stream);
  (void)hipMemsetAsync(c2, 0, 16 * 512 * 4, stream);
  const size_t dec_lds = 16 * 2048 * 4 + 16 * 1024 * 2;
  for (int t = 0; t < T; ++t)
    lstm_step_dec_k<<<1, 512, dec_lds, stream>>>(W1h, W2h, gx1, gx2, db1, db2,
        h1_16, h1f, c1, h2f, c2, pin16, zh1, zc1, zh2, zc2, t, T);

  // ---- projection -> mel_pred (first half of d_out)
  float* outp = (float*)d_out;
  float* out2 = outp + (size_t)BT * MEL;
  gemmT(pin16, 1536, Projw, 80, outp, 80, BT, 80, 1536, prb, 0);

  // ---- postnet: 4x(conv k5 + BN + tanh) + final conv, residual add
  im2col_k<<<blks((long long)BT * 416), 256, 0, stream>>>(outp, A16, B, T, 80, 5, 416);
  gemmT(A16, 416, Post0, 512, ppA, 512, BT, 512, 416, nullptr, 0);
  bn_stats_k<<<512, 256, 0, stream>>>(ppA, mean, var, BT, 512);
  bn_act_k<<<blks((long long)BT * 512), 256, 0, stream>>>(ppA, ppB, mean, var, pobs, pobo, BT, 512, 2);
  for (int i = 0; i < 3; ++i) {
    im2col_k<<<blks((long long)BT * 2560), 256, 0, stream>>>(ppB, A16, B, T, 512, 5, 2560);
    gemmT(A16, 2560, Postm[i], 512, ppA, 512, BT, 512, 2560, nullptr, 0);
    bn_stats_k<<<512, 256, 0, stream>>>(ppA, mean, var, BT, 512);
    bn_act_k<<<blks((long long)BT * 512), 256, 0, stream>>>(ppA, ppB, mean, var,
        pobs + (i + 1) * 512, pobo + (i + 1) * 512, BT, 512, 2);
  }
  im2col_k<<<blks((long long)BT * 2560), 256, 0, stream>>>(ppB, A16, B, T, 512, 5, 2560);
  gemmT(A16, 2560, Post4, 80, resb, 80, BT, 80, 2560, nullptr, 0);
  add_out_k<<<blks((long long)BT * MEL), 256, 0, stream>>>(outp, resb, pob4, out2, BT * MEL, MEL);
}
